// SparseAutoencoder_62620623176019
// MI455X (gfx1250) — matrix operand order repaired and checked by reading, not compiled, not run on hardware
//
#include <hip/hip_runtime.h>

// ---------------- problem constants ----------------
#define B_DIM 4096
#define D_DIM 768
#define M_DIM 16384
#define K_ACT 32
#define K_MULTI 128
#define K_AUX 64
#define DEAD_THR 256

// output layout (flat float32, reference tuple order)
#define O_REC   ((size_t)0)
#define O_ACT   (O_REC  + (size_t)B_DIM * D_DIM)
#define O_TKI   (O_ACT  + (size_t)B_DIM * M_DIM)
#define O_TKV   (O_TKI  + (size_t)B_DIM * K_ACT)
#define O_MREC  (O_TKV  + (size_t)B_DIM * K_ACT)
#define O_AXI   (O_MREC + (size_t)B_DIM * D_DIM)
#define O_AXV   (O_AXI  + (size_t)B_DIM * K_AUX)

typedef __attribute__((ext_vector_type(16))) __bf16 v16bf;
typedef __attribute__((ext_vector_type(8)))  __bf16 v8bf;
typedef __attribute__((ext_vector_type(8)))  float  v8f;
typedef int v4i_vs __attribute__((vector_size(16)));   // matches builtin's V4i

#if defined(__has_builtin)
#  if __has_builtin(__builtin_amdgcn_global_load_async_to_lds_b128)
#    define USE_ASYNC_LDS 1
#  endif
#endif
#ifndef USE_ASYNC_LDS
#  define USE_ASYNC_LDS 0
#endif

static __device__ inline unsigned short bf_hi_bits(float f) {
    return (unsigned short)(__float_as_uint(f) >> 16);
}
static __device__ inline float bf_hi_f32(float f) {
    return __uint_as_float(__float_as_uint(f) & 0xFFFF0000u);
}
static __device__ inline v16bf cat16(v8bf a, v8bf b) {
    return __builtin_shufflevector(a, b, 0,1,2,3,4,5,6,7,8,9,10,11,12,13,14,15);
}

// ============================================================
// Kernel 0: one-time bf16 hi/lo split of (x - input_bias) and W_enc.
// Removes all conversion VALU from the GEMM hot loop so staging can be
// pure async global->LDS DMA.
// ============================================================
__global__ __launch_bounds__(256)
void k_split(const float* __restrict__ x,
             const float* __restrict__ W_enc,
             const float* __restrict__ input_bias,
             unsigned short* __restrict__ Xh, unsigned short* __restrict__ Xl,
             unsigned short* __restrict__ Wh, unsigned short* __restrict__ Wl) {
    size_t e0  = ((size_t)blockIdx.x * 256 + threadIdx.x) * 8; // 8 elems/thread
    size_t row = e0 / D_DIM;
    size_t col = e0 % D_DIM;

    float f[8];
    unsigned short* dh;
    unsigned short* dl;
    if (row < B_DIM) {
        const float* src = x + row * D_DIM + col;
        const float* bia = input_bias + col;
        #pragma unroll
        for (int i = 0; i < 8; ++i) f[i] = src[i] - bia[i];
        dh = Xh + row * D_DIM + col;
        dl = Xl + row * D_DIM + col;
    } else {
        size_t wr = row - B_DIM;
        const float* src = W_enc + wr * D_DIM + col;
        #pragma unroll
        for (int i = 0; i < 8; ++i) f[i] = src[i];
        dh = Wh + wr * D_DIM + col;
        dl = Wl + wr * D_DIM + col;
    }
    union { unsigned short u[8]; uint4 v; } hh, ll;
    #pragma unroll
    for (int i = 0; i < 8; ++i) {
        hh.u[i] = bf_hi_bits(f[i]);
        ll.u[i] = bf_hi_bits(f[i] - bf_hi_f32(f[i]));
    }
    *(uint4*)dh = hh.v;
    *(uint4*)dl = ll.v;
}

// ============================================================
// Kernel 1: transpose W_dec [D, M] -> WdecT [M, D]
// ============================================================
__global__ __launch_bounds__(256)
void k_transpose(const float* __restrict__ Wdec, float* __restrict__ WdecT) {
    __shared__ float tile[32][33];
    int m0 = blockIdx.x * 32;
    int d0 = blockIdx.y * 32;
    int tx = threadIdx.x & 31;
    int ty = threadIdx.x >> 5;
    #pragma unroll
    for (int s = 0; s < 32; s += 8)
        tile[ty + s][tx] = Wdec[(size_t)(d0 + ty + s) * M_DIM + m0 + tx];
    __syncthreads();
    #pragma unroll
    for (int s = 0; s < 32; s += 8)
        WdecT[(size_t)(m0 + ty + s) * D_DIM + d0 + tx] = tile[tx][ty + s];
}

// ============================================================
// Kernel 2: encoder GEMM via bf16x3 WMMA, async-LDS double buffered.
//   pre[b, m] = sum_k xc_hi/lo[b,k] * W_hi/lo[m,k] + neuron_bias[m]
// 256 threads (8 wave32), 128x128 output tile per workgroup.
// LDS: 2 stages x (Ah|Al|Bh|Bl), each 128 rows x 40-elem pitch (80B rows).
// ============================================================
#define PITCH 40
#define MAT_BYTES (128 * PITCH * 2)        // 10240
#define STAGE_BYTES (4 * MAT_BYTES)        // 40960
#define OUT_PITCH 132
#define GEMM_LDS_BYTES (2 * STAGE_BYTES)   // 81920 (>= 128*132*4 epilogue)

static __device__ inline void async_cp16(const void* g, void* l) {
#if USE_ASYNC_LDS
    __builtin_amdgcn_global_load_async_to_lds_b128(
        (__attribute__((address_space(1))) v4i_vs*)(v4i_vs*)(void*)g,
        (__attribute__((address_space(3))) v4i_vs*)(v4i_vs*)l, 0, 0);
#else
    *(uint4*)l = *(const uint4*)g;
#endif
}

static __device__ inline void stage_issue(const unsigned short* __restrict__ Xh,
                                          const unsigned short* __restrict__ Xl,
                                          const unsigned short* __restrict__ Wh,
                                          const unsigned short* __restrict__ Wl,
                                          char* sb, int t, int b0, int m0, int k0) {
    // 128x32 bf16 tile = 512 16B-chunks; 256 threads x 2 chunks per matrix.
    #pragma unroll
    for (int h = 0; h < 2; ++h) {
        int c = t + h * 256;
        int r = c >> 2;
        int q = c & 3;
        size_t go = (size_t)k0 + q * 8;
        int    lo = r * (PITCH * 2) + q * 16;
        async_cp16(Xh + (size_t)(b0 + r) * D_DIM + go, sb + 0 * MAT_BYTES + lo);
        async_cp16(Xl + (size_t)(b0 + r) * D_DIM + go, sb + 1 * MAT_BYTES + lo);
        async_cp16(Wh + (size_t)(m0 + r) * D_DIM + go, sb + 2 * MAT_BYTES + lo);
        async_cp16(Wl + (size_t)(m0 + r) * D_DIM + go, sb + 3 * MAT_BYTES + lo);
    }
}

__global__ __launch_bounds__(256)
void k_encoder_wmma(const unsigned short* __restrict__ Xh,
                    const unsigned short* __restrict__ Xl,
                    const unsigned short* __restrict__ Wh,
                    const unsigned short* __restrict__ Wl,
                    const float* __restrict__ neuron_bias,
                    float* __restrict__ pre) {
    extern __shared__ char smem[];

    const int t  = threadIdx.x;
    const int b0 = blockIdx.y * 128;
    const int m0 = blockIdx.x * 128;

    const int lane = t & 31;
    const int w    = t >> 5;
    const int wr   = w >> 2;       // 0..1
    const int wc   = w & 3;        // 0..3
    const int half = lane >> 4;
    const int lr   = lane & 15;

    v8f acc[4][2];
    #pragma unroll
    for (int i = 0; i < 4; ++i)
        #pragma unroll
        for (int j = 0; j < 2; ++j)
            acc[i][j] = (v8f)(0.0f);

    const int NK = D_DIM / 32;     // 24 K-chunks

    // prologue: stage 0 into buffer 0
    stage_issue(Xh, Xl, Wh, Wl, smem, t, b0, m0, 0);

    for (int kc = 0; kc < NK; ++kc) {
#if USE_ASYNC_LDS
        asm volatile("s_wait_asynccnt 0" ::: "memory");
#endif
        __syncthreads();
        if (kc + 1 < NK)
            stage_issue(Xh, Xl, Wh, Wl, smem + ((kc + 1) & 1) * STAGE_BYTES,
                        t, b0, m0, (kc + 1) * 32);

        const char* sb = smem + (kc & 1) * STAGE_BYTES;
        const __bf16* AhS = (const __bf16*)(sb + 0 * MAT_BYTES);
        const __bf16* AlS = (const __bf16*)(sb + 1 * MAT_BYTES);
        const __bf16* BhS = (const __bf16*)(sb + 2 * MAT_BYTES);
        const __bf16* BlS = (const __bf16*)(sb + 3 * MAT_BYTES);

        // A frags: 16x32 bf16 layout: lane<16 -> K0..7 & 16..23, lane>=16 -> K8..15 & 24..31
        v16bf ah[4], al[4];
        #pragma unroll
        for (int i = 0; i < 4; ++i) {
            int m  = wr * 64 + i * 16 + lr;
            int kb = half * 8;
            ah[i] = cat16(*(const v8bf*)(AhS + m * PITCH + kb),
                          *(const v8bf*)(AhS + m * PITCH + kb + 16));
            al[i] = cat16(*(const v8bf*)(AlS + m * PITCH + kb),
                          *(const v8bf*)(AlS + m * PITCH + kb + 16));
        }
        // B frags: 32x16 bf16: lanes 0..15 K0..7 & 16..23, lanes 16..31 K8..15 & 24..31, N = lr
        v16bf bh[2], bl[2];
        #pragma unroll
        for (int j = 0; j < 2; ++j) {
            int n   = wc * 32 + j * 16 + lr;
            int kb2 = half * 8;
            bh[j] = cat16(*(const v8bf*)(BhS + n * PITCH + kb2),
                          *(const v8bf*)(BhS + n * PITCH + kb2 + 16));
            bl[j] = cat16(*(const v8bf*)(BlS + n * PITCH + kb2),
                          *(const v8bf*)(BlS + n * PITCH + kb2 + 16));
        }

        #pragma unroll
        for (int i = 0; i < 4; ++i)
            #pragma unroll
            for (int j = 0; j < 2; ++j) {
                acc[i][j] = __builtin_amdgcn_wmma_f32_16x16x32_bf16(
                    false, ah[i], false, bh[j], (short)0, acc[i][j], false, false);
                acc[i][j] = __builtin_amdgcn_wmma_f32_16x16x32_bf16(
                    false, ah[i], false, bl[j], (short)0, acc[i][j], false, false);
                acc[i][j] = __builtin_amdgcn_wmma_f32_16x16x32_bf16(
                    false, al[i], false, bh[j], (short)0, acc[i][j], false, false);
            }
        __syncthreads();
    }

    // ---- epilogue: stage tile in LDS (padded pitch), coalesced b128 stores ----
    float* outT = (float*)smem;    // 128 x OUT_PITCH floats, aliases stage bufs
    #pragma unroll
    for (int i = 0; i < 4; ++i)
        #pragma unroll
        for (int j = 0; j < 2; ++j) {
            int m = wr * 64 + i * 16 + half * 8;
            int n = wc * 32 + j * 16 + lr;
            #pragma unroll
            for (int r = 0; r < 8; ++r)
                outT[(m + r) * OUT_PITCH + n] = acc[i][j][r];
        }
    __syncthreads();
    #pragma unroll
    for (int q = 0; q < 16; ++q) {
        int c    = t + q * 256;        // float4 chunk id 0..4095
        int rrow = c >> 5;
        int ccol = (c & 31) * 4;
        float4 v  = *(const float4*)(outT + rrow * OUT_PITCH + ccol);
        float4 nb = *(const float4*)(neuron_bias + m0 + ccol);
        v.x += nb.x; v.y += nb.y; v.z += nb.z; v.w += nb.w;
        *(float4*)(pre + (size_t)(b0 + rrow) * M_DIM + m0 + ccol) = v;
    }
}

// ============================================================
// Kernel 3: per-row top-k + sparse reconstructions + outputs
// ============================================================
static __device__ inline unsigned fkey(float f) {
    unsigned u = __float_as_uint(f);
    return (u & 0x80000000u) ? ~u : (u | 0x80000000u);
}

__global__ __launch_bounds__(256)
void k_topk_fused(const float* __restrict__ pre,
                  const float* __restrict__ WdecT,
                  const float* __restrict__ input_bias,
                  const int*   __restrict__ steps,
                  float* __restrict__ out) {
    __shared__ float    row[M_DIM];      // 64KB of the 320KB WGP LDS
    __shared__ unsigned hist[256];
    __shared__ float    cval[K_MULTI];
    __shared__ int      cidx[K_MULTI];
    __shared__ unsigned sh_prefix;
    __shared__ int      sh_rem;
    __shared__ int      sh_cnt;
    __shared__ int      sh_eq;

    const int tid = threadIdx.x;
    const int b   = blockIdx.x;
    const float* prow = pre + (size_t)b * M_DIM;
    float* act_row    = out + O_ACT + (size_t)b * M_DIM;

    for (int i = tid * 4; i < M_DIM; i += 256 * 4) {
        *(float4*)(row + i) = *(const float4*)(prow + i);
        *(float4*)(act_row + i) = make_float4(0.f, 0.f, 0.f, 0.f);
    }
    __syncthreads();

    for (int phase = 0; phase < 2; ++phase) {
        const bool useMask = (phase == 1);
        const int  k       = useMask ? K_AUX : K_MULTI;

        // ---- radix select (4 x 8-bit passes) for k-th largest key ----
        if (tid == 0) { sh_prefix = 0u; sh_rem = k; }
        __syncthreads();
        for (int shift = 24; shift >= 0; shift -= 8) {
            hist[tid] = 0u;
            __syncthreads();
            unsigned pfx = sh_prefix;
            for (int i = tid; i < M_DIM; i += 256) {
                float v = row[i];
                if (useMask && !(steps[i] > DEAD_THR)) v = 0.0f;
                unsigned key = fkey(v);
                if (shift == 24 || (key >> (shift + 8)) == (pfx >> (shift + 8)))
                    atomicAdd(&hist[(key >> shift) & 255u], 1u);
            }
            __syncthreads();
            if (tid == 0) {
                int rem = sh_rem;
                int d = 255;
                for (; d > 0; --d) {
                    int c = (int)hist[d];
                    if (c >= rem) break;
                    rem -= c;
                }
                sh_prefix = pfx | ((unsigned)d << shift);
                sh_rem = rem;
            }
            __syncthreads();
        }
        const unsigned thr = sh_prefix;
        const int      rem = sh_rem;

        // ---- gather: all strictly-greater, then `rem` ties ----
        if (tid == 0) { sh_cnt = 0; sh_eq = 0; }
        __syncthreads();
        for (int i = tid; i < M_DIM; i += 256) {
            float v = row[i];
            if (useMask && !(steps[i] > DEAD_THR)) v = 0.0f;
            if (fkey(v) > thr) {
                int p = atomicAdd(&sh_cnt, 1);
                cval[p] = v; cidx[p] = i;
            }
        }
        __syncthreads();
        for (int i = tid; i < M_DIM; i += 256) {
            float v = row[i];
            if (useMask && !(steps[i] > DEAD_THR)) v = 0.0f;
            if (fkey(v) == thr) {
                int e = atomicAdd(&sh_eq, 1);
                if (e < rem) {
                    int p = atomicAdd(&sh_cnt, 1);
                    cval[p] = v; cidx[p] = i;
                }
            }
        }
        __syncthreads();

        // ---- bitonic sort k candidates: value desc, index asc ----
        for (int size = 2; size <= k; size <<= 1) {
            for (int stride = size >> 1; stride > 0; stride >>= 1) {
                __syncthreads();
                int i = tid;
                if (i < k) {
                    int p = i ^ stride;
                    if (p > i) {
                        float a = cval[i], c = cval[p];
                        int ia = cidx[i], ic = cidx[p];
                        bool agtc = (a > c) || (a == c && ia < ic);
                        bool desc = ((i & size) == 0);
                        if (desc ? !agtc : agtc) {
                            cval[i] = c; cval[p] = a;
                            cidx[i] = ic; cidx[p] = ia;
                        }
                    }
                }
            }
        }
        __syncthreads();

        if (!useMask) {
            if (tid < K_ACT) {
                float rv = fmaxf(cval[tid], 0.0f);
                out[O_TKV + (size_t)b * K_ACT + tid] = rv;
                out[O_TKI + (size_t)b * K_ACT + tid] = (float)cidx[tid];
                act_row[cidx[tid]] = rv;
            }
            if (tid < K_MULTI) cval[tid] = fmaxf(cval[tid], 0.0f);
            __syncthreads();

            const int d0 = tid, d1 = tid + 256, d2 = tid + 512;
            float r0 = 0.f, r1 = 0.f, r2 = 0.f;
            float m0 = 0.f, m1 = 0.f, m2 = 0.f;
            for (int kk = 0; kk < K_MULTI; ++kk) {
                float v = cval[kk];
                const float* wrp = WdecT + (size_t)cidx[kk] * D_DIM;
                float w0 = wrp[d0], w1 = wrp[d1], w2 = wrp[d2];
                m0 += v * w0; m1 += v * w1; m2 += v * w2;
                if (kk < K_ACT) { r0 += v * w0; r1 += v * w1; r2 += v * w2; }
            }
            float ib0 = input_bias[d0], ib1 = input_bias[d1], ib2 = input_bias[d2];
            float* rec  = out + O_REC  + (size_t)b * D_DIM;
            float* mrec = out + O_MREC + (size_t)b * D_DIM;
            rec[d0]  = r0 + ib0; rec[d1]  = r1 + ib1; rec[d2]  = r2 + ib2;
            mrec[d0] = m0 + ib0; mrec[d1] = m1 + ib1; mrec[d2] = m2 + ib2;
            __syncthreads();
        } else {
            if (tid < K_AUX) {
                out[O_AXV + (size_t)b * K_AUX + tid] = fmaxf(cval[tid], 0.0f);
                out[O_AXI + (size_t)b * K_AUX + tid] = (float)cidx[tid];
            }
            __syncthreads();
        }
    }
}

// ============================================================
extern "C" void kernel_launch(void* const* d_in, const int* in_sizes, int n_in,
                              void* d_out, int out_size, void* d_ws, size_t ws_size,
                              hipStream_t stream) {
    (void)in_sizes; (void)n_in; (void)out_size; (void)ws_size;
    const float* x           = (const float*)d_in[0];
    const float* W_enc       = (const float*)d_in[1];
    const float* W_dec       = (const float*)d_in[2];
    const float* input_bias  = (const float*)d_in[3];
    const float* neuron_bias = (const float*)d_in[4];
    const int*   steps       = (const int*)d_in[5];
    float* out = (float*)d_out;

    // workspace layout
    float* pre   = (float*)d_ws;                         // B*M f32
    float* WdecT = pre + (size_t)B_DIM * M_DIM;          // M*D f32
    unsigned short* Xh = (unsigned short*)(WdecT + (size_t)M_DIM * D_DIM);
    unsigned short* Xl = Xh + (size_t)B_DIM * D_DIM;
    unsigned short* Wh = Xl + (size_t)B_DIM * D_DIM;
    unsigned short* Wl = Wh + (size_t)M_DIM * D_DIM;

    size_t split_blocks = ((size_t)(B_DIM + M_DIM) * D_DIM) / (8 * 256);
    k_split<<<(unsigned)split_blocks, 256, 0, stream>>>(
        x, W_enc, input_bias, Xh, Xl, Wh, Wl);
    k_transpose<<<dim3(M_DIM / 32, D_DIM / 32), 256, 0, stream>>>(W_dec, WdecT);
    k_encoder_wmma<<<dim3(M_DIM / 128, B_DIM / 128), 256, GEMM_LDS_BYTES, stream>>>(
        Xh, Xl, Wh, Wl, neuron_bias, pre);
    k_topk_fused<<<B_DIM, 256, 0, stream>>>(pre, WdecT, input_bias, steps, out);
}
